// TPUGraphNetwork_46256797778158
// MI455X (gfx1250) — matrix operand order repaired and checked by reading, not compiled, not run on hardware
//
#include <hip/hip_runtime.h>

typedef __attribute__((ext_vector_type(16))) _Float16 v16h;
typedef __attribute__((ext_vector_type(8)))  _Float16 v8h;
typedef __attribute__((ext_vector_type(8)))  float    v8f;

#define LN_EPS 1e-5f

// ---------------------------------------------------------------- utilities
__global__ void zero_f32(float* p, long long n) {
  long long i = (long long)blockIdx.x * blockDim.x + threadIdx.x;
  if (i < n) p[i] = 0.0f;
}

__global__ void deg_count(const int* __restrict__ rows, float* __restrict__ deg, int E) {
  int e = blockIdx.x * blockDim.x + threadIdx.x;
  if (e < E) atomicAdd(&deg[rows[e]], 1.0f);
}

__global__ void edge_vals(const int* __restrict__ rows, const float* __restrict__ deg,
                          float* __restrict__ vals, int E) {
  int e = blockIdx.x * blockDim.x + threadIdx.x;
  if (e < E) vals[e] = 1.0f / fmaxf(deg[rows[e]], 1.0f);
}

// Swizzle f32 weights [layers][srcRows][128] into fragment-major f16:
// dst[l][(kc*8+tc)*32 + lane][16 halves], lane = half*16 + (c&15),
// half = (k%32)/16, j = k%16. Rows k >= srcRows are zero-padded (K multiple of 32).
__global__ void prep_w16_swz(const float* __restrict__ src, _Float16* __restrict__ dst,
                             int srcRows, int K, int layers) {
  int i = blockIdx.x * blockDim.x + threadIdx.x;
  int perL = K * 128;
  if (i >= layers * perL) return;
  int l = i / perL, r = i - l * perL;
  int k = r >> 7, c = r & 127;
  float v = (k < srcRows) ? src[((size_t)l * srcRows + k) * 128 + c] : 0.0f;
  int kc = k >> 5, kl = k & 31;
  int half = kl >> 4, j = kl & 15;
  int tc = c >> 4, ln = c & 15;
  int lane = half * 16 + ln;
  dst[(size_t)l * perL + (((size_t)(kc * 8 + tc) * 32 + lane) << 4) + j] = (_Float16)v;
}

// Build padded f16 input rows: [emb(op) | features[1:128] | 0] -> K=160 cols.
__global__ void build_xin(const float* __restrict__ features, const float* __restrict__ emb,
                          _Float16* __restrict__ xin, int F, int EMBD, int K) {
  long long idx = blockIdx.x;          // node index in [0, B*N)
  int t = threadIdx.x;                 // 0..K-1 (blockDim = 160)
  if (t >= K) return;
  const float* frow = features + idx * F;
  float v;
  if (t < EMBD) {
    int op = (int)frow[0];
    op = op < 0 ? 0 : (op > 127 ? 127 : op);
    v = emb[op * EMBD + t];
  } else {
    int f = t - EMBD + 1;              // 1..127 valid, 128 -> pad
    v = (f < F) ? frow[f] : 0.0f;
  }
  xin[idx * K + t] = (_Float16)v;
}

// Per-edge gather/scale/scatter: agg[b,row,:] += x[b,col,:] * val  (f32 atomics, L2-resident)
__global__ void scatter_msgs(const float* __restrict__ x, const float* __restrict__ vals,
                             const int* __restrict__ rows, const int* __restrict__ cols,
                             float* __restrict__ agg, int N) {
  int e = blockIdx.x;
  int t = threadIdx.x;                 // 256 threads: b = t>>7, c = t&127
  int b = t >> 7, c = t & 127;
  int r = rows[e], cl = cols[e];
  float v = vals[e];
  atomicAdd(&agg[((long long)b * N + r) * 128 + c],
            x[((long long)b * N + cl) * 128 + c] * v);
}

// ---------------------------------------------------------------- fused WMMA GEMM + SiLU + LN
// A: [nrows, K] (f16 or f32) row-major, Wg: pre-swizzled fragment-major f16,
// out: [nrows, 128] f32. 128 threads = 4 waves; each wave: 16 rows x 128 cols.
template<int K, bool A_F32>
__global__ __launch_bounds__(128)
void gemm_silu_ln(const void* __restrict__ Aptr, const _Float16* __restrict__ Wg,
                  const float* __restrict__ bias, const float* __restrict__ gamma,
                  const float* __restrict__ beta, float* __restrict__ Xout, int nrows) {
  constexpr int ROWS = 64;
  constexpr int KC = K / 32;
  constexpr int AB_BYTES  = (ROWS * K + K * 128) * 2;   // f16 A frags + f16 W frags
  constexpr int OUT_BYTES = ROWS * 128 * 4;             // f32 silu output tile
  constexpr int SMEM = AB_BYTES > OUT_BYTES ? AB_BYTES : OUT_BYTES;
  __shared__ __align__(16) char smem[SMEM];             // A/W region reused for output
  __shared__ float sBias[128], sG[128], sBeta[128];
  _Float16* sA = (_Float16*)smem;                       // [ROWS/16 * KC][32 lanes][16]
  _Float16* sW = sA + ROWS * K;                         // [KC*8 tiles][32 lanes][16]
  float*    sO = (float*)smem;                          // [ROWS][128] (after barrier)

  const int t = threadIdx.x;
  const long long rows0 = (long long)blockIdx.x * ROWS;

  sBias[t] = bias[t]; sG[t] = gamma[t]; sBeta[t] = beta[t];

  // stage W: plain contiguous vector copy (already fragment-major in global)
  {
    const uint4* wg4 = (const uint4*)Wg;
    uint4* sw4 = (uint4*)sW;
    for (int i = t; i < K * 128 / 8; i += 128) sw4[i] = wg4[i];
  }
  // stage A: each thread handles one (row, group-of-8 k). A group of 8 consecutive
  // k maps to 8 contiguous halves inside one lane's fragment slice -> b128 LDS store.
  for (int i = t; i < ROWS * (K / 8); i += 128) {
    int r = i / (K / 8);
    int k = (i - r * (K / 8)) * 8;
    int kc = k >> 5, kl = k & 31;
    int half = kl >> 4;
    int jbase = kl & 15;
    int lane = half * 16 + (r & 15);
    int wv = r >> 4;
    _Float16* dst = sA + ((((wv * KC + kc) * 32 + lane) << 4) + jbase);
    long long gr = rows0 + r;
    v8h hv;
    if (gr < nrows) {
      if (A_F32) {
        const float* src = (const float*)Aptr + gr * K + k;
#pragma unroll
        for (int j = 0; j < 8; ++j) hv[j] = (_Float16)src[j];
      } else {
        hv = *(const v8h*)((const _Float16*)Aptr + gr * K + k);
      }
    } else {
#pragma unroll
      for (int j = 0; j < 8; ++j) hv[j] = (_Float16)0.0f;
    }
    *(v8h*)dst = hv;
  }
  __syncthreads();

  const int wave = t >> 5, lane = t & 31;
  const int half = lane >> 4, ln = lane & 15;

  v8f acc[8] = {};
#pragma unroll
  for (int kc = 0; kc < KC; ++kc) {
    const v16h a = *(const v16h*)(sA + (((wave * KC + kc) * 32 + lane) << 4));
#pragma unroll
    for (int tc = 0; tc < 8; ++tc) {
      const v16h bfr = *(const v16h*)(sW + (((kc * 8 + tc) * 32 + lane) << 4));
      acc[tc] = __builtin_amdgcn_wmma_f32_16x16x32_f16(
          false, a, false, bfr, (short)0, acc[tc], false, false);
    }
  }
  __syncthreads();  // all waves done reading sA/sW -> safe to overwrite as sO

  // bias + SiLU into LDS (C layout: VGPR r -> row half*8+r, col = tile*16+ln)
#pragma unroll
  for (int tc = 0; tc < 8; ++tc) {
    const int cc = tc * 16 + ln;
#pragma unroll
    for (int r = 0; r < 8; ++r) {
      int mr = wave * 16 + half * 8 + r;
      float y = acc[tc][r] + sBias[cc];
      sO[mr * 128 + cc] = y * (1.0f / (1.0f + __expf(-y)));
    }
  }
  __syncthreads();

  // LayerNorm: each wave normalizes its own 16 rows, wave32 shuffle reduction.
  for (int rr = 0; rr < 16; ++rr) {
    const int mrow = wave * 16 + rr;
    const long long grow = rows0 + mrow;
    float v0 = sO[mrow * 128 + lane],      v1 = sO[mrow * 128 + lane + 32];
    float v2 = sO[mrow * 128 + lane + 64], v3 = sO[mrow * 128 + lane + 96];
    float s  = v0 + v1 + v2 + v3;
    float s2 = v0 * v0 + v1 * v1 + v2 * v2 + v3 * v3;
#pragma unroll
    for (int off = 16; off > 0; off >>= 1) {
      s  += __shfl_xor(s,  off, 32);
      s2 += __shfl_xor(s2, off, 32);
    }
    float mu   = s  * (1.0f / 128.0f);
    float var  = s2 * (1.0f / 128.0f) - mu * mu;
    float rstd = rsqrtf(var + LN_EPS);
    if (grow < nrows) {
      Xout[grow * 128 + lane]      = (v0 - mu) * rstd * sG[lane]      + sBeta[lane];
      Xout[grow * 128 + lane + 32] = (v1 - mu) * rstd * sG[lane + 32] + sBeta[lane + 32];
      Xout[grow * 128 + lane + 64] = (v2 - mu) * rstd * sG[lane + 64] + sBeta[lane + 64];
      Xout[grow * 128 + lane + 96] = (v3 - mu) * rstd * sG[lane + 96] + sBeta[lane + 96];
    }
  }
}

// ---------------------------------------------------------------- ragged pooling + projection
__global__ void pool_proj(const float* __restrict__ x, const int* __restrict__ lengths,
                          const float* __restrict__ wproj, const float* __restrict__ bproj,
                          float* __restrict__ out, int N, int B, int G) {
  int g = blockIdx.x / B;
  int b = blockIdx.x % B;
  int t = threadIdx.x;                 // 256 threads
  int start = 0;
  for (int i = 0; i < g; ++i) start += lengths[i];
  int len = lengths[g];
  int c = t & 127, hfl = t >> 7;
  float acc = 0.0f;
  for (int n = start + hfl; n < start + len; n += 2)
    acc += x[((long long)b * N + n) * 128 + c];
  __shared__ float red[256];
  red[t] = acc; __syncthreads();
  float p = 0.0f;
  if (t < 128) p = (red[t] + red[t + 128]) * wproj[c];
  __syncthreads(); red[t] = p; __syncthreads();
  for (int off = 64; off > 0; off >>= 1) {
    if (t < off) red[t] += red[t + off];
    __syncthreads();
  }
  if (t == 0) out[g * B + b] = red[0] + bproj[0];
}

// ---------------------------------------------------------------- launch
extern "C" void kernel_launch(void* const* d_in, const int* in_sizes, int n_in,
                              void* d_out, int out_size, void* d_ws, size_t ws_size,
                              hipStream_t stream) {
  const float* features = (const float*)d_in[0];
  const int*   edge_idx = (const int*)d_in[1];
  const int*   lengths  = (const int*)d_in[2];
  const float* emb      = (const float*)d_in[3];
  const float* W_in     = (const float*)d_in[4];
  const float* b_in     = (const float*)d_in[5];
  const float* g_in     = (const float*)d_in[6];
  const float* beta_in  = (const float*)d_in[7];
  const float* msg_W    = (const float*)d_in[8];
  const float* msg_b    = (const float*)d_in[9];
  const float* msg_g    = (const float*)d_in[10];
  const float* msg_beta = (const float*)d_in[11];
  const float* W_proj   = (const float*)d_in[12];
  const float* b_proj   = (const float*)d_in[13];
  float* out = (float*)d_out;

  const int F = 128, C = 128, EMB = 32, B = 2, KIN = 160;
  const int E  = in_sizes[1] / 2;
  const int G  = in_sizes[2];
  const int N  = in_sizes[0] / (B * F);
  const int BN = B * N;

  const int* rows = edge_idx;
  const int* cols = edge_idx + E;

  // workspace carve-out (256B aligned)
  char* ws = (char*)d_ws;
  size_t off = 0;
  auto carve = [&](size_t bytes) -> void* {
    void* p = ws + off;
    off += (bytes + 255) & ~(size_t)255;
    return p;
  };
  float*    deg    = (float*)   carve((size_t)N * 4);
  float*    vals   = (float*)   carve((size_t)E * 4);
  _Float16* w16in  = (_Float16*)carve((size_t)KIN * C * 2);
  _Float16* w16msg = (_Float16*)carve((size_t)3 * C * C * 2);
  _Float16* xin    = (_Float16*)carve((size_t)BN * KIN * 2);
  float*    xbuf   = (float*)   carve((size_t)BN * C * 4);
  float*    agg    = (float*)   carve((size_t)BN * C * 4);

  // 1) degree-normalized edge values
  zero_f32<<<(N + 255) / 256, 256, 0, stream>>>(deg, N);
  deg_count<<<(E + 255) / 256, 256, 0, stream>>>(rows, deg, E);
  edge_vals<<<(E + 255) / 256, 256, 0, stream>>>(rows, deg, vals, E);

  // 2) f16 weight prep (fragment-major swizzle)
  prep_w16_swz<<<(KIN * C + 255) / 256, 256, 0, stream>>>(W_in, w16in, F - 1 + EMB, KIN, 1);
  prep_w16_swz<<<(3 * C * C + 255) / 256, 256, 0, stream>>>(msg_W, w16msg, C, C, 3);

  // 3) input layer: embed+concat -> padded f16 rows, then fused WMMA GEMM+SiLU+LN
  build_xin<<<BN, 160, 0, stream>>>(features, emb, xin, F, EMB, KIN);
  gemm_silu_ln<KIN, false><<<(BN + 63) / 64, 128, 0, stream>>>(
      xin, w16in, b_in, g_in, beta_in, xbuf, BN);

  // 4) three message-passing layers
  const long long aggN = (long long)BN * C;
  for (int l = 0; l < 3; ++l) {
    zero_f32<<<(int)((aggN + 255) / 256), 256, 0, stream>>>(agg, aggN);
    scatter_msgs<<<E, 256, 0, stream>>>(xbuf, vals, rows, cols, agg, N);
    gemm_silu_ln<C, true><<<(BN + 63) / 64, 128, 0, stream>>>(
        agg, w16msg + (size_t)l * C * C, msg_b + l * C, msg_g + l * C,
        msg_beta + l * C, xbuf, BN);
  }

  // 5) ragged pool + projection
  pool_proj<<<G * B, 256, 0, stream>>>(xbuf, lengths, W_proj, b_proj, out, N, B, G);
}
